// KsModel_44049184588156
// MI455X (gfx1250) — matrix operand order repaired and checked by reading, not compiled, not run on hardware
//
#include <hip/hip_runtime.h>
#include <hip/hip_bf16.h>
#include <math.h>

typedef __attribute__((ext_vector_type(16))) __bf16 v16bf;
typedef __attribute__((ext_vector_type(8)))  __bf16 v8bf;
typedef __attribute__((ext_vector_type(8)))  float  v8f;

#define BS    8
#define NNODE 1024
#define FIN   768
#define DHEAD 128
#define NHEAD 6
#define HID   768
#define ALPHA 0.2f
#define NEGV  (-9.0e15f)
#define APAD  72   // padded row/col stride (elements) for LDS bank spread, 16B-aligned

__device__ __forceinline__ float leakyf(float v) { return v >= 0.f ? v : ALPHA * v; }

__device__ __forceinline__ v16bf cat16(v8bf lo, v8bf hi) {
  return __builtin_shufflevector(lo, hi, 0,1,2,3,4,5,6,7,8,9,10,11,12,13,14,15);
}

// ---------------- elementwise converts ----------------
__global__ void k_cast_bf16(const float* __restrict__ src, __bf16* __restrict__ dst, int n) {
  int i = blockIdx.x * blockDim.x + threadIdx.x;
  if (i < n) dst[i] = (__bf16)src[i];
}

// W: [H, FIN, DHEAD] -> dst[fin, h*DHEAD + d]   (FIN x HID)
__global__ void k_pack_W(const float* __restrict__ W, __bf16* __restrict__ dst) {
  int i = blockIdx.x * blockDim.x + threadIdx.x;
  if (i >= FIN * HID) return;
  int col = i % HID, fin = i / HID;
  int h = col / DHEAD, d = col % DHEAD;
  dst[i] = (__bf16)W[(h * FIN + fin) * DHEAD + d];
}

// ---------------- generic bf16 WMMA GEMM ----------------
// Block = 256 threads (8 waves), tile 16 rows x 128 cols, K-step 64 (2 WMMA/wave/step).
// mode 0: store bf16. mode 1: store f32 + bias[col]. reluA: relu on A operand.
__global__ void k_gemm(const __bf16* __restrict__ A, const __bf16* __restrict__ B,
                       void* __restrict__ out, const float* __restrict__ bias,
                       int K, int Nn, long strideA, long strideOut,
                       int mode, int reluA) {
  __shared__ __attribute__((aligned(16))) __bf16 As[16 * APAD];    // row-major, padded
  __shared__ __attribute__((aligned(16))) __bf16 BsT[128 * APAD];  // [col][k], padded
  int tid  = threadIdx.x;
  int wave = tid >> 5, lane = tid & 31;
  int i0 = blockIdx.x * 16;
  int colBase = blockIdx.y * 128;
  long aOff = (long)blockIdx.z * strideA;
  long oOff = (long)blockIdx.z * strideOut;
  v8f acc = {};
  int n = lane & 15;
  int mrow = lane & 15;
  int half = (lane < 16) ? 0 : 1;

  for (int k0 = 0; k0 < K; k0 += 64) {
    __syncthreads();
    // A tile 16x64 via uint4 (8 bf16) vector loads; relu applied packed
    if (tid < 128) {
      int r = tid >> 3, q = tid & 7;
      union { uint4 u; __bf16 h[8]; } v;
      v.u = *(const uint4*)(A + aOff + (long)(i0 + r) * K + k0 + q * 8);
      if (reluA) {
        #pragma unroll
        for (int j = 0; j < 8; ++j) if ((float)v.h[j] < 0.f) v.h[j] = (__bf16)0.f;
      }
      *(uint4*)(As + r * APAD + q * 8) = v.u;
    }
    // B tile 64x128 -> transposed into BsT[col][k] (conflict-free b16 store pattern)
    for (int idx = tid; idx < 1024; idx += 256) {
      int r = idx & 63, c8 = idx >> 6;
      union { uint4 u; __bf16 h[8]; } v;
      v.u = *(const uint4*)(B + (long)(k0 + r) * Nn + colBase + c8 * 8);
      #pragma unroll
      for (int j = 0; j < 8; ++j) BsT[(c8 * 8 + j) * APAD + r] = v.h[j];
    }
    if (k0 + 64 < K) {   // global_prefetch_b8 next K-tile
      __builtin_prefetch(B + (long)(k0 + 64 + (tid & 63)) * Nn + colBase, 0, 0);
      if (tid < 16)
        __builtin_prefetch(A + aOff + (long)(i0 + tid) * K + k0 + 64, 0, 0);
    }
    __syncthreads();

    #pragma unroll
    for (int s = 0; s < 2; ++s) {
      // A fragment: two ds_load_b128 (per-lane K runs are contiguous)
      const __bf16* ap = As + mrow * APAD + s * 32 + half * 8;
      v16bf af = cat16(*(const v8bf*)ap, *(const v8bf*)(ap + 16));
      // B fragment: two ds_load_b128 from column-major tile
      const __bf16* bp = BsT + (wave * 16 + n) * APAD + s * 32 + half * 8;
      v16bf bf = cat16(*(const v8bf*)bp, *(const v8bf*)(bp + 16));
      acc = __builtin_amdgcn_wmma_f32_16x16x32_bf16(false, af, false, bf,
                                                    (short)0, acc, false, false);
    }
  }

  int col = colBase + wave * 16 + n;
  #pragma unroll
  for (int r = 0; r < 8; ++r) {
    int mi = r + half * 8;
    long idx = oOff + (long)(i0 + mi) * Nn + col;
    float v = acc[r];
    if (mode == 1) ((float*)out)[idx] = v + bias[col];
    else           ((__bf16*)out)[idx] = (__bf16)v;
  }
}

// ---------------- f/g head projections ----------------
__global__ void k_dot_fg(const __bf16* __restrict__ X, const float* __restrict__ a1,
                         const float* __restrict__ a2, float* __restrict__ f,
                         float* __restrict__ g, int nH, int Dh) {
  int idx = blockIdx.x * blockDim.x + threadIdx.x;
  if (idx >= BS * nH * NNODE) return;
  int i = idx % NNODE; int bh = idx / NNODE; int b = bh / nH; int h = bh % nH;
  const __bf16* row = X + (long)(b * NNODE + i) * HID + h * Dh;
  const float* w1 = a1 + h * Dh;
  const float* w2 = a2 + h * Dh;
  float accf = 0.f, accg = 0.f;
  for (int d8 = 0; d8 < Dh; d8 += 8) {
    union { uint4 u; __bf16 h[8]; } v;
    v.u = *(const uint4*)(row + d8);
    #pragma unroll
    for (int j = 0; j < 8; ++j) {
      float w = (float)v.h[j];
      accf += w * w1[d8 + j]; accg += w * w2[d8 + j];
    }
  }
  f[idx] = accf; g[idx] = accg;
}

// ---------------- column softmax stats (softmax over i) ----------------
__global__ void k_colstats(const int* __restrict__ adj, const float* __restrict__ f,
                           const float* __restrict__ g, float* __restrict__ cmax,
                           float* __restrict__ csum, int nH) {
  int idx = blockIdx.x * blockDim.x + threadIdx.x;
  if (idx >= BS * nH * NNODE) return;
  int j = idx % NNODE; int bh = idx / NNODE; int b = bh / nH;
  const int* acol = adj + (long)b * NNODE * NNODE + j;
  const float* fr = f + (long)bh * NNODE;
  float gj = g[idx];
  float m = -INFINITY, s = 0.f;
  for (int i = 0; i < NNODE; ++i) {
    float v = (acol[(long)i * NNODE] > 0) ? leakyf(fr[i] + gj) : NEGV;
    if (v > m) { s *= __expf(m - v); m = v; }
    s += __expf(v - m);
  }
  cmax[idx] = m; csum[idx] = s;
}

// ---------------- fused  elu( softmax(att) @ Wh )  WMMA kernel ----------------
// A fragment rebuilt on the fly from f/g/colstats + adj; B tile staged transposed.
// mode 0: bf16 out (layer1 concat). mode 1: f32 out (layer2 z).
__global__ void k_fused_att(const int* __restrict__ adj, const float* __restrict__ f,
                            const float* __restrict__ g, const float* __restrict__ cmax,
                            const float* __restrict__ csum, const __bf16* __restrict__ Bmat,
                            void* __restrict__ out, int nH, int headCols, int mode) {
  __shared__ __attribute__((aligned(16))) int    adjS[16 * 64];
  __shared__ __attribute__((aligned(16))) __bf16 BsT[128 * APAD];
  __shared__ float gS[64], mS[64], sS[64], fS[16];
  int tid  = threadIdx.x;
  int wave = tid >> 5, lane = tid & 31;
  int zz = blockIdx.z;
  int b = zz / nH;
  int statsOff = zz * NNODE;
  int colBase = (zz % nH) * headCols + blockIdx.y * 128;
  int i0 = blockIdx.x * 16;
  long bOff = (long)b * NNODE * HID;
  long adjOff = (long)b * NNODE * NNODE;
  if (tid < 16) fS[tid] = f[statsOff + i0 + tid];
  v8f acc = {};
  int n = lane & 15;
  int mrow = lane & 15;
  int half = (lane < 16) ? 0 : 1;

  for (int j0 = 0; j0 < NNODE; j0 += 64) {
    __syncthreads();
    {   // adj tile 16x64 ints as uint4
      int r = tid >> 4, c4 = tid & 15;
      *(uint4*)(adjS + r * 64 + c4 * 4) =
          *(const uint4*)(adj + adjOff + (long)(i0 + r) * NNODE + j0 + c4 * 4);
    }
    if (tid < 64) {
      gS[tid] = g[statsOff + j0 + tid];
      mS[tid] = cmax[statsOff + j0 + tid];
      sS[tid] = csum[statsOff + j0 + tid];
    }
    for (int idx = tid; idx < 1024; idx += 256) {   // Wh tile 64x128 -> transposed
      int r = idx & 63, c8 = idx >> 6;
      union { uint4 u; __bf16 h[8]; } v;
      v.u = *(const uint4*)(Bmat + bOff + (long)(j0 + r) * HID + colBase + c8 * 8);
      #pragma unroll
      for (int j = 0; j < 8; ++j) BsT[(c8 * 8 + j) * APAD + r] = v.h[j];
    }
    if (j0 + 64 < NNODE) {
      __builtin_prefetch(Bmat + bOff + (long)(j0 + 64 + (tid & 63)) * HID + colBase, 0, 0);
      if (tid < 16)
        __builtin_prefetch(adj + adjOff + (long)(i0 + tid) * NNODE + j0 + 64, 0, 0);
    }
    __syncthreads();

    float fi = fS[mrow];
    #pragma unroll
    for (int s = 0; s < 2; ++s) {
      v16bf af;
      #pragma unroll
      for (int t = 0; t < 16; ++t) {
        int kk = s * 32 + half * 8 + ((t >= 8) ? 16 : 0) + (t & 7);
        float v = (adjS[mrow * 64 + kk] > 0) ? leakyf(fi + gS[kk]) : NEGV;
        af[t] = (__bf16)(__expf(v - mS[kk]) / sS[kk]);
      }
      const __bf16* bp = BsT + (wave * 16 + n) * APAD + s * 32 + half * 8;
      v16bf bf = cat16(*(const v8bf*)bp, *(const v8bf*)(bp + 16));
      acc = __builtin_amdgcn_wmma_f32_16x16x32_bf16(false, af, false, bf,
                                                    (short)0, acc, false, false);
    }
  }

  int col = colBase + wave * 16 + n;
  #pragma unroll
  for (int r = 0; r < 8; ++r) {
    int mi = r + half * 8;
    float v = acc[r];
    v = v > 0.f ? v : (__expf(v) - 1.f);        // ELU
    long idx = (long)b * NNODE * HID + (long)(i0 + mi) * HID + col;
    if (mode == 1) ((float*)out)[idx] = v;
    else           ((__bf16*)out)[idx] = (__bf16)v;
  }
}

// ---------------- per-(b,d) logsumexp over i, folded into C[b] ----------------
__global__ void k_col_lse(const float* __restrict__ z, const float* __restrict__ kw,
                          float* __restrict__ C) {
  int idx = blockIdx.x * blockDim.x + threadIdx.x;
  if (idx >= BS * HID) return;
  int b = idx / HID, d = idx % HID;
  const float* col = z + (long)b * NNODE * HID + d;
  float m = -INFINITY, s = 0.f;
  for (int i = 0; i < NNODE; ++i) {
    float v = col[(long)i * HID];
    if (v > m) { s *= __expf(m - v); m = v; }
    s += __expf(v - m);
  }
  atomicAdd(&C[b], (m + __logf(s)) * kw[d]);
}

// ---------------- final: know = (proj+z)@kw + kb - C[b], masked ----------------
__global__ void k_final(const float* __restrict__ proj, const float* __restrict__ z,
                        const float* __restrict__ kw, const float* __restrict__ kb,
                        const float* __restrict__ C, const int* __restrict__ nm,
                        float* __restrict__ out) {
  __shared__ float red[256];
  int row = blockIdx.x;   // b*NNODE + i
  int tid = threadIdx.x;
  const float* pr = proj + (long)row * HID;
  const float* zr = z    + (long)row * HID;
  float acc = 0.f;
  for (int d = tid; d < HID; d += 256) acc += (pr[d] + zr[d]) * kw[d];
  red[tid] = acc;
  __syncthreads();
  for (int s = 128; s > 0; s >>= 1) {
    if (tid < s) red[tid] += red[tid + s];
    __syncthreads();
  }
  if (tid == 0) {
    float r = red[0] + kb[0] - C[row / NNODE];
    out[row] = (nm[row] < 1) ? -INFINITY : r;
  }
}

extern "C" void kernel_launch(void* const* d_in, const int* in_sizes, int n_in,
                              void* d_out, int out_size, void* d_ws, size_t ws_size,
                              hipStream_t stream) {
  (void)in_sizes; (void)n_in; (void)out_size; (void)ws_size;
  const float* x   = (const float*)d_in[0];
  const int*   adj = (const int*)d_in[1];
  const int*   nm  = (const int*)d_in[2];
  const float* W   = (const float*)d_in[3];
  const float* a1  = (const float*)d_in[4];
  const float* a2  = (const float*)d_in[5];
  const float* Wo  = (const float*)d_in[6];
  const float* ao1 = (const float*)d_in[7];
  const float* ao2 = (const float*)d_in[8];
  const float* Wp  = (const float*)d_in[9];
  const float* bp  = (const float*)d_in[10];
  const float* kw  = (const float*)d_in[11];
  const float* kb  = (const float*)d_in[12];
  float* out = (float*)d_out;

  char* base = (char*)d_ws;
  size_t off = 0;
  auto alloc = [&](size_t bytes) -> void* {
    void* p = base + off;
    off = (off + bytes + 255) & ~(size_t)255;
    return p;
  };
  __bf16* Xbf    = (__bf16*)alloc((size_t)BS * NNODE * FIN * 2);
  __bf16* Wcatbf = (__bf16*)alloc((size_t)FIN * HID * 2);
  __bf16* Wobf   = (__bf16*)alloc((size_t)HID * HID * 2);
  __bf16* Wpbf   = (__bf16*)alloc((size_t)FIN * HID * 2);
  __bf16* Whbf   = (__bf16*)alloc((size_t)BS * NNODE * HID * 2);
  __bf16* x2bf   = (__bf16*)alloc((size_t)BS * NNODE * HID * 2);
  __bf16* Whobf  = (__bf16*)alloc((size_t)BS * NNODE * HID * 2);
  float* f1  = (float*)alloc((size_t)BS * NHEAD * NNODE * 4);
  float* g1  = (float*)alloc((size_t)BS * NHEAD * NNODE * 4);
  float* cm1 = (float*)alloc((size_t)BS * NHEAD * NNODE * 4);
  float* cs1 = (float*)alloc((size_t)BS * NHEAD * NNODE * 4);
  float* f2  = (float*)alloc((size_t)BS * NNODE * 4);
  float* g2  = (float*)alloc((size_t)BS * NNODE * 4);
  float* cm2 = (float*)alloc((size_t)BS * NNODE * 4);
  float* cs2 = (float*)alloc((size_t)BS * NNODE * 4);
  float* zbuf = (float*)alloc((size_t)BS * NNODE * HID * 4);
  float* proj = (float*)alloc((size_t)BS * NNODE * HID * 4);
  float* Cb   = (float*)alloc((size_t)BS * 4);

  const int T = 256;
  hipMemsetAsync(Cb, 0, BS * sizeof(float), stream);

  k_cast_bf16<<<(BS * NNODE * FIN + T - 1) / T, T, 0, stream>>>(x, Xbf, BS * NNODE * FIN);
  k_pack_W   <<<(FIN * HID + T - 1) / T,        T, 0, stream>>>(W, Wcatbf);
  k_cast_bf16<<<(HID * HID + T - 1) / T,        T, 0, stream>>>(Wo, Wobf, HID * HID);
  k_cast_bf16<<<(FIN * HID + T - 1) / T,        T, 0, stream>>>(Wp, Wpbf, FIN * HID);

  dim3 gg(NNODE / 16, HID / 128, BS);
  // GEMM1: Wh = X @ Wcat  (bf16 out)
  k_gemm<<<gg, T, 0, stream>>>(Xbf, Wcatbf, Whbf, nullptr, FIN, HID,
                               (long)NNODE * FIN, (long)NNODE * HID, 0, 0);
  k_dot_fg  <<<(BS * NHEAD * NNODE + T - 1) / T, T, 0, stream>>>(Whbf, a1, a2, f1, g1, NHEAD, DHEAD);
  k_colstats<<<(BS * NHEAD * NNODE + T - 1) / T, T, 0, stream>>>(adj, f1, g1, cm1, cs1, NHEAD);
  // fused layer-1 attention: x2 = elu(att @ Wh), heads concatenated
  k_fused_att<<<dim3(NNODE / 16, 1, BS * NHEAD), T, 0, stream>>>(
      adj, f1, g1, cm1, cs1, Whbf, x2bf, NHEAD, DHEAD, 0);
  // GEMM2: Who = x2 @ Wo
  k_gemm<<<gg, T, 0, stream>>>(x2bf, Wobf, Whobf, nullptr, HID, HID,
                               (long)NNODE * HID, (long)NNODE * HID, 0, 0);
  k_dot_fg  <<<(BS * NNODE + T - 1) / T, T, 0, stream>>>(Whobf, ao1, ao2, f2, g2, 1, HID);
  k_colstats<<<(BS * NNODE + T - 1) / T, T, 0, stream>>>(adj, f2, g2, cm2, cs2, 1);
  // fused out-layer attention: z = elu(atto @ Who)  (f32 out)
  k_fused_att<<<dim3(NNODE / 16, HID / 128, BS), T, 0, stream>>>(
      adj, f2, g2, cm2, cs2, Whobf, zbuf, 1, HID, 1);
  k_col_lse<<<(BS * HID + T - 1) / T, T, 0, stream>>>(zbuf, kw, Cb);
  // proj = relu(X) @ Wp + bp  (f32 out with bias, relu fused into A load)
  k_gemm<<<gg, T, 0, stream>>>(Xbf, Wpbf, proj, bp, FIN, HID,
                               (long)NNODE * FIN, (long)NNODE * HID, 1, 1);
  k_final<<<BS * NNODE, T, 0, stream>>>(proj, zbuf, kw, kb, Cb, nm, out);
}
